// CNNVectorForm_34668976014087
// MI455X (gfx1250) — matrix operand order repaired and checked by reading, not compiled, not run on hardware
//
#include <hip/hip_runtime.h>
#include <math.h>

typedef __attribute__((ext_vector_type(16))) _Float16 v16h;
typedef __attribute__((ext_vector_type(8)))  _Float16 v8h;
typedef __attribute__((ext_vector_type(8)))  float    v8f;

union AOp { v16h v; v8h h[2]; };

#define WMMA_F16(a, b, c) \
  __builtin_amdgcn_wmma_f32_16x16x32_f16(false, (a), false, (b), (short)0, (c), false, false)

// ---------------- problem constants ----------------
#define BATCH   8192
#define IMG_H   28
#define IMG_W   28
#define CONV_K  25          // 5x5
#define COUT    20
#define POOLED  2880        // 20*12*12
#define FC1_OUT 500
#define FC1_PAD 512
#define FC2_OUT 10

#define IMG_STRIDE 29       // LDS row stride (f16 elems), odd to spread banks
#define IMG_ELEMS  (IMG_H * IMG_STRIDE)

// ---------------- workspace layout (bytes) ----------------
static constexpr size_t OFF_CONVB = 0;                       //  2048 B
static constexpr size_t OFF_FC2B  = 4096;                    // 16384 B
static constexpr size_t OFF_FC1B  = 32768;                   // ~2.95 MB
static constexpr size_t OFF_ACT1  = 4u * 1024u * 1024u;      // 47.2 MB
static constexpr size_t OFF_ACT2  = 52u * 1024u * 1024u;     //  8.4 MB

// =========================================================
// Weight packing into the WMMA B-operand register image:
// element e (0..15) of lane l in tile (kt, nt) holds
//   B[k][n]  with  k = kt*32 + (e>>3)*16 + (l>>4)*8 + (e&7),  n = nt*16 + (l&15)
// Padded K rows / N cols are ZERO, which also lets the GEMMs feed
// arbitrary finite garbage in the corresponding A elements.
// =========================================================
__global__ void pack_convB(const float* __restrict__ w, _Float16* __restrict__ dst) {
  int idx = blockIdx.x * blockDim.x + threadIdx.x;       // 2*32*16 = 1024
  if (idx >= 2 * 32 * 16) return;
  int e = idx & 15, l = (idx >> 4) & 31, t = idx >> 9;
  int oc = t * 16 + (l & 15);
  int k  = (e >> 3) * 16 + (l >> 4) * 8 + (e & 7);
  float v = 0.f;
  if (oc < COUT && k < CONV_K) v = w[oc * CONV_K + k];
  dst[idx] = (_Float16)v;
}

__global__ void pack_fc1B(const float* __restrict__ w, _Float16* __restrict__ dst) {
  int total = 90 * 32 * 32 * 16;
  for (int idx = blockIdx.x * blockDim.x + threadIdx.x; idx < total;
       idx += gridDim.x * blockDim.x) {
    int e  = idx & 15;
    int l  = (idx >> 4) & 31;
    int nt = (idx >> 9) & 31;
    int kt = idx >> 14;
    int n = nt * 16 + (l & 15);
    int k = kt * 32 + (e >> 3) * 16 + (l >> 4) * 8 + (e & 7);
    float v = 0.f;
    if (n < FC1_OUT) v = w[(size_t)n * POOLED + k];      // k < 2880 always
    dst[idx] = (_Float16)v;
  }
}

__global__ void pack_fc2B(const float* __restrict__ w, _Float16* __restrict__ dst) {
  int idx = blockIdx.x * blockDim.x + threadIdx.x;       // 16*32*16 = 8192
  if (idx >= 16 * 32 * 16) return;
  int e  = idx & 15;
  int l  = (idx >> 4) & 31;
  int kt = idx >> 9;
  int n = l & 15;
  int k = kt * 32 + (e >> 3) * 16 + (l >> 4) * 8 + (e & 7);
  float v = 0.f;
  if (n < FC2_OUT && k < FC1_OUT) v = w[n * FC1_OUT + k];
  dst[idx] = (_Float16)v;
}

// =========================================================
// Fused conv(5x5, VALID) + bias + 2x2 maxpool.
// Block = 2 images, 4 waves per image (9 position-tiles each).
// im2col GEMM per tile: C[pos16 x oc16] = A[pos16 x k32] * B[k32 x oc16].
// Position tile = 2 rows x 8 cols: m -> (py = r0+(m&1), px = c0+(m>>1)),
// so each lane's 8 C values pool (in groups of 4) into two 2x2 pool outputs.
// A elements with K >= 25 load finite garbage (offset 0); B rows K>=25 are 0.
// =========================================================
__global__ __launch_bounds__(256) void conv_pool_kernel(
    const float* __restrict__ x, const float* __restrict__ conv_b,
    const _Float16* __restrict__ convB, _Float16* __restrict__ act1) {
  __shared__ _Float16 simg[2][IMG_ELEMS];

  const int lane = threadIdx.x & 31;
  const int wave = threadIdx.x >> 5;

  // ---- cooperative staging of 2 images (f32 -> f16), coalesced
  {
    const float* xb = x + (size_t)blockIdx.x * 2 * (IMG_H * IMG_W);
    for (int idx = threadIdx.x; idx < 2 * IMG_H * IMG_W; idx += 256) {
      int il  = (idx >= IMG_H * IMG_W) ? 1 : 0;
      int rem = idx - il * (IMG_H * IMG_W);
      int r   = rem / IMG_W;
      int c   = rem - r * IMG_W;
      simg[il][r * IMG_STRIDE + c] = (_Float16)xb[idx];
    }
  }
  __syncthreads();

  const int il   = wave >> 2;               // image within block
  const int sub  = wave & 3;                // tile-range within image
  const int img  = blockIdx.x * 2 + il;

  // ---- B tiles (weights): one 32B load per lane per tile
  v16h b0 = *(const v16h*)(convB + (size_t)(0 * 32 + lane) * 16);
  v16h b1 = *(const v16h*)(convB + (size_t)(1 * 32 + lane) * 16);

  const int m  = lane & 15;
  const int hi = lane >> 4;
  const int py_off = m & 1;
  const int px_off = m >> 1;

  const int   oc0   = m;                    // < 16, always valid
  const int   oc1   = 16 + m;               // valid only if m < 4
  const float bias0 = conv_b[oc0];
  const float bias1 = (m < 4) ? conv_b[oc1] : 0.f;

  // ---- A-assembly offsets (ISA 16-bit A 16x32 layout):
  // element e of lane holds K = ((e>>1)&4 ? 16:0) + hi*8 + ((e>>1)&3)*2 + (e&1)
  // invalid K (>= 25) -> load in-bounds garbage at offset 0 (B row is zero)
  int offs[16];
#pragma unroll
  for (int e = 0; e < 16; ++e) {
    int j = e >> 1;
    int K = ((j & 4) ? 16 : 0) + hi * 8 + (j & 3) * 2 + (e & 1);
    if (K < CONV_K) {
      int ky = K / 5, kx = K - ky * 5;
      offs[e] = ky * IMG_STRIDE + kx;
    } else {
      offs[e] = 0;
    }
  }

  _Float16* aout = act1 + (size_t)img * POOLED;

  for (int i = 0; i < 9; ++i) {
    const int t  = sub * 9 + i;
    const int r0 = (t / 3) * 2;             // output row pair
    const int c0 = (t % 3) * 8;             // output col group
    const _Float16* base = &simg[il][(r0 + py_off) * IMG_STRIDE + (c0 + px_off)];

    v16h a;
#pragma unroll
    for (int e = 0; e < 16; ++e) a[e] = base[offs[e]];   // unconditional gathers

    v8f c0acc = {};
    c0acc = WMMA_F16(a, b0, c0acc);
    v8f c1acc = {};
    c1acc = WMMA_F16(a, b1, c1acc);

    // ---- pool 2x2 in registers (C layout: lane = col, VGPR v = row v + hi*8)
    const int pr  = r0 >> 1;
    const int pcb = (c0 >> 1) + hi * 2;

    float p0 = fmaxf(fmaxf(c0acc[0], c0acc[1]), fmaxf(c0acc[2], c0acc[3])) + bias0;
    float p1 = fmaxf(fmaxf(c0acc[4], c0acc[5]), fmaxf(c0acc[6], c0acc[7])) + bias0;
    _Float16* o0 = aout + oc0 * 144 + pr * 12 + pcb;
    o0[0] = (_Float16)p0;
    o0[1] = (_Float16)p1;

    if (m < 4) {
      float q0 = fmaxf(fmaxf(c1acc[0], c1acc[1]), fmaxf(c1acc[2], c1acc[3])) + bias1;
      float q1 = fmaxf(fmaxf(c1acc[4], c1acc[5]), fmaxf(c1acc[6], c1acc[7])) + bias1;
      _Float16* o1 = aout + oc1 * 144 + pr * 12 + pcb;
      o1[0] = (_Float16)q0;
      o1[1] = (_Float16)q1;
    }
  }
}

// =========================================================
// FC1: [8192 x 2880] x [2880 x 512(500)] + bias, ReLU -> act2 f16 [8192 x 512]
// Block = 64 batch rows (4 M-subtiles), 8 waves x 4 N-tiles = 512 cols.
// A operand = two contiguous b128 loads per lane (ISA A layout halves).
// B registers are reused across the 4 M-subtiles -> 16 WMMAs per K-step.
// =========================================================
__global__ __launch_bounds__(256) void fc1_kernel(
    const _Float16* __restrict__ act1, const _Float16* __restrict__ fc1B,
    const float* __restrict__ fc1_b, _Float16* __restrict__ act2) {
  const int lane = threadIdx.x & 31;
  const int wave = threadIdx.x >> 5;
  const int mbase = blockIdx.x * 64;
  const int m  = lane & 15;
  const int hi = lane >> 4;
  const int hi8 = hi * 8;

  const _Float16* arow0 = act1 + (size_t)(mbase + m) * POOLED;

  v8f acc[4][4] = {};                       // [m-subtile][n-subtile]
  for (int kt = 0; kt < 90; ++kt) {
    const int kbase = kt * 32;

    v16h b[4];
#pragma unroll
    for (int i = 0; i < 4; ++i) {
      int nt = wave * 4 + i;
      b[i] = *(const v16h*)(fc1B + (((size_t)kt * 32 + nt) * 32 + lane) * 16);
    }

#pragma unroll
    for (int mt = 0; mt < 4; ++mt) {
      const _Float16* arow = arow0 + (size_t)mt * 16 * POOLED;
      AOp a;
      a.h[0] = *(const v8h*)(arow + kbase + hi8);        // K = kbase+hi*8 .. +7
      a.h[1] = *(const v8h*)(arow + kbase + 16 + hi8);   // K = kbase+16+hi*8 .. +7
#pragma unroll
      for (int i = 0; i < 4; ++i) acc[mt][i] = WMMA_F16(a.v, b[i], acc[mt][i]);
    }
  }

  // epilogue: bias + ReLU, store f16 (padded cols 500..511 come out as 0)
#pragma unroll
  for (int i = 0; i < 4; ++i) {
    int n = (wave * 4 + i) * 16 + m;
    float bias = (n < FC1_OUT) ? fc1_b[n] : 0.f;
#pragma unroll
    for (int mt = 0; mt < 4; ++mt) {
#pragma unroll
      for (int v = 0; v < 8; ++v) {
        int row = mbase + mt * 16 + v + hi8;
        float val = fmaxf(acc[mt][i][v] + bias, 0.f);
        act2[(size_t)row * FC1_PAD + n] = (_Float16)val;
      }
    }
  }
}

// =========================================================
// FC2 + softmax: [8192 x 512] x [512 x 16(10)] -> softmax over 10 -> f32 out
// One wave per 16-row tile; row-wise softmax via shfl_xor within 16-lane groups.
// =========================================================
__global__ __launch_bounds__(256) void fc2_softmax_kernel(
    const _Float16* __restrict__ act2, const _Float16* __restrict__ fc2B,
    const float* __restrict__ fc2_b, float* __restrict__ out) {
  const int lane = threadIdx.x & 31;
  const int wave = threadIdx.x >> 5;
  const int mbase = (blockIdx.x * 8 + wave) * 16;
  const int m  = lane & 15;
  const int hi = lane >> 4;
  const int hi8 = hi * 8;

  const _Float16* arow = act2 + (size_t)(mbase + m) * FC1_PAD;

  v8f acc = {};
  for (int kt = 0; kt < 16; ++kt) {
    const int kbase = kt * 32;
    AOp a;
    a.h[0] = *(const v8h*)(arow + kbase + hi8);
    a.h[1] = *(const v8h*)(arow + kbase + 16 + hi8);
    v16h b = *(const v16h*)(fc2B + ((size_t)kt * 32 + lane) * 16);
    acc = WMMA_F16(a.v, b, acc);
  }

  const float bias = (m < FC2_OUT) ? fc2_b[m] : 0.f;
#pragma unroll
  for (int v = 0; v < 8; ++v) {
    float val = (m < FC2_OUT) ? (acc[v] + bias) : -__builtin_inff();
    float r = val;
    r = fmaxf(r, __shfl_xor(r, 8, 16));
    r = fmaxf(r, __shfl_xor(r, 4, 16));
    r = fmaxf(r, __shfl_xor(r, 2, 16));
    r = fmaxf(r, __shfl_xor(r, 1, 16));
    float ex = (m < FC2_OUT) ? expf(val - r) : 0.f;
    float s = ex;
    s += __shfl_xor(s, 8, 16);
    s += __shfl_xor(s, 4, 16);
    s += __shfl_xor(s, 2, 16);
    s += __shfl_xor(s, 1, 16);
    if (m < FC2_OUT) {
      int row = mbase + v + hi8;
      out[(size_t)row * FC2_OUT + m] = ex / s;
    }
  }
}

// =========================================================
extern "C" void kernel_launch(void* const* d_in, const int* in_sizes, int n_in,
                              void* d_out, int out_size, void* d_ws, size_t ws_size,
                              hipStream_t stream) {
  (void)in_sizes; (void)n_in; (void)out_size; (void)ws_size;

  const float* x      = (const float*)d_in[0];
  const float* conv_w = (const float*)d_in[1];
  const float* conv_b = (const float*)d_in[2];
  const float* fc1_w  = (const float*)d_in[3];
  const float* fc1_b  = (const float*)d_in[4];
  const float* fc2_w  = (const float*)d_in[5];
  const float* fc2_b  = (const float*)d_in[6];
  float* out = (float*)d_out;

  char* ws = (char*)d_ws;
  _Float16* convB = (_Float16*)(ws + OFF_CONVB);
  _Float16* fc2B  = (_Float16*)(ws + OFF_FC2B);
  _Float16* fc1B  = (_Float16*)(ws + OFF_FC1B);
  _Float16* act1  = (_Float16*)(ws + OFF_ACT1);
  _Float16* act2  = (_Float16*)(ws + OFF_ACT2);

  // 1) pack weights into WMMA B-operand register images
  pack_convB<<<4, 256, 0, stream>>>(conv_w, convB);
  pack_fc1B<<<2048, 256, 0, stream>>>(fc1_w, fc1B);
  pack_fc2B<<<32, 256, 0, stream>>>(fc2_w, fc2B);

  // 2) fused conv + bias + maxpool  (2 images per block, 4 waves per image)
  conv_pool_kernel<<<BATCH / 2, 256, 0, stream>>>(x, conv_b, convB, act1);

  // 3) FC1 + ReLU  (64-row M-tile per block)
  fc1_kernel<<<BATCH / 64, 256, 0, stream>>>(act1, fc1B, fc1_b, act2);

  // 4) FC2 + softmax
  fc2_softmax_kernel<<<BATCH / (16 * 8), 256, 0, stream>>>(act2, fc2B, fc2_b, out);
}
